// TrillNet_19310172963036
// MI455X (gfx1250) — matrix operand order repaired and checked by reading, not compiled, not run on hardware
//
#include <hip/hip_runtime.h>
#include <math.h>

// ---------------- problem constants ----------------
namespace cfg {
constexpr int B    = 8;
constexpr int HT   = 16;
constexpr int S    = 64;
constexpr int DIM  = 128;
constexpr int C    = 2000;
constexpr int D    = 256;          // 2*DIM
constexpr int NH   = 4;
constexpr int HD   = 64;           // D / NH
constexpr int WIN  = 4;
constexpr int L    = 2;
constexpr int DFF  = 1024;         // 4*D
constexpr int NSEQ = B * HT + B;   // 136 (history seqs + current seqs share encoder)
constexpr int ROWS = NSEQ * S;     // 8704
constexpr int EC   = B * HT * (S - 1); // 8064 edges
}

typedef __bf16 bf16_t;
typedef __attribute__((ext_vector_type(16))) __bf16 v16bf;
typedef __attribute__((ext_vector_type(8)))  float  v8f;
typedef __attribute__((ext_vector_type(4)))  int    v4i;

// =====================================================================
// Generic bf16 WMMA GEMM:  C[M,N] = act( A[M,K] @ B[K,N] + bias + residual )
// B is passed PRE-TRANSPOSED as Bt[N][K] so both tiles stage with
// contiguous 16B chunks (no LDS software transpose).
// block = 128 threads (4 wave32s, 2x2 wave grid), block tile 64x64, BK=32.
// Requires K % 32 == 0 and N % 16 == 0 (true for all call sites).
// =====================================================================
#define GEMM_BM 64
#define GEMM_BN 64
#define GEMM_BK 32
#define LDS_PAD 8

#if defined(__gfx1250__)
#  if __has_builtin(__builtin_amdgcn_global_load_async_to_lds_b128)
#    define GEMM_ASYNC 1
#  endif
#endif

#if defined(GEMM_ASYNC)
// 16-byte global->LDS async copy (GLOBAL_LOAD_ASYNC_TO_LDS_B128, ASYNCcnt).
// Builtin takes v4i pointers in AS1 (global) / AS3 (LDS); integer round-trips
// keep the address-space casts legal in C++.
#define ASYNC_CP16(g, l, off)                                                        \
    __builtin_amdgcn_global_load_async_to_lds_b128(                                  \
        (__attribute__((address_space(1))) v4i*)(unsigned long long)(g),             \
        (__attribute__((address_space(3))) v4i*)(unsigned)(unsigned long long)(l),   \
        off, 0)
#endif

__global__ __launch_bounds__(128)
void k_gemm_bf16(const bf16_t* __restrict__ A, const bf16_t* __restrict__ Bt,
                 float* __restrict__ Cout,
                 const float* __restrict__ bias,      // per-col (N), may be null
                 const float* __restrict__ residual,  // MxN, may be null
                 int M, int N, int K, int relu_flag)
{
    __shared__ bf16_t sA[GEMM_BM][GEMM_BK + LDS_PAD];  // (m,k)
    __shared__ bf16_t sB[GEMM_BN][GEMM_BK + LDS_PAD];  // (n,k)

    const int tid  = threadIdx.x;
    const int wave = tid >> 5;
    const int lane = tid & 31;
    const int wm   = wave >> 1;   // 0..1
    const int wn   = wave & 1;    // 0..1
    const int block_m = blockIdx.y * GEMM_BM;
    const int block_n = blockIdx.x * GEMM_BN;
    const int row16 = tid >> 1;          // 0..63 : tile row for staging
    const int koff  = (tid & 1) * 16;    // 0 / 16 : k-chunk for staging

    const bool full = (block_m + GEMM_BM <= M) && (block_n + GEMM_BN <= N);

    v8f acc[2][2];
#pragma unroll
    for (int im = 0; im < 2; ++im)
#pragma unroll
        for (int in = 0; in < 2; ++in)
#pragma unroll
            for (int e = 0; e < 8; ++e) acc[im][in][e] = 0.0f;

    for (int k0 = 0; k0 < K; k0 += GEMM_BK) {
        const bf16_t* ga = A  + (size_t)(block_m + row16) * K + k0 + koff;
        const bf16_t* gb = Bt + (size_t)(block_n + row16) * K + k0 + koff;
        bf16_t* la = &sA[row16][koff];
        bf16_t* lb = &sB[row16][koff];

        if (full) {
#if defined(GEMM_ASYNC)
            ASYNC_CP16(ga, la, 0);  ASYNC_CP16(ga, la, 16);
            ASYNC_CP16(gb, lb, 0);  ASYNC_CP16(gb, lb, 16);
#else
            const uint4* pa = (const uint4*)ga;
            const uint4* pb = (const uint4*)gb;
            uint4 a0 = pa[0], a1 = pa[1], b0 = pb[0], b1 = pb[1];
            ((uint4*)la)[0] = a0; ((uint4*)la)[1] = a1;
            ((uint4*)lb)[0] = b0; ((uint4*)lb)[1] = b1;
#endif
            if (k0 + GEMM_BK < K) {                    // global_prefetch_b8
                __builtin_prefetch(ga + GEMM_BK, 0, 0);
                __builtin_prefetch(gb + GEMM_BK, 0, 0);
            }
        } else {
            // edge tiles: masked vector path with zero fill (K always in-bounds)
            const uint4 z = {0u, 0u, 0u, 0u};
            uint4 a0 = z, a1 = z, b0 = z, b1 = z;
            if (block_m + row16 < M) { const uint4* pa = (const uint4*)ga; a0 = pa[0]; a1 = pa[1]; }
            if (block_n + row16 < N) { const uint4* pb = (const uint4*)gb; b0 = pb[0]; b1 = pb[1]; }
            ((uint4*)la)[0] = a0; ((uint4*)la)[1] = a1;
            ((uint4*)lb)[0] = b0; ((uint4*)lb)[1] = b1;
        }
#if defined(GEMM_ASYNC)
        if (full) {
#  if __has_builtin(__builtin_amdgcn_s_wait_asynccnt)
            __builtin_amdgcn_s_wait_asynccnt(0);
#  else
            asm volatile("s_wait_asynccnt 0" ::: "memory");
#  endif
        }
#endif
        __syncthreads();

        // ---- fragments + WMMA (2x2 16x16 tiles per wave over K=32 slab) ----
        const int kbb = (lane < 16) ? 0 : 16;   // B layout: elems 0-15 -> K kbb..+15
        v16bf bfr[2];
#pragma unroll
        for (int in = 0; in < 2; ++in) {
            const int col = wn * 32 + in * 16 + (lane & 15);
            const bf16_t* bpp = &sB[col][kbb];
#pragma unroll
            for (int e = 0; e < 16; ++e) bfr[in][e] = bpp[e];
        }
        const int kba0 = (lane < 16) ? 0 : 16;  // A layout: elems 0-15 -> K kba0..+15
#pragma unroll
        for (int im = 0; im < 2; ++im) {
            const int row = wm * 32 + im * 16 + (lane & 15);
            const bf16_t* ap = &sA[row][0];
            v16bf afr;
#pragma unroll
            for (int e = 0; e < 8; ++e) afr[e]     = ap[kba0 + e];
#pragma unroll
            for (int e = 0; e < 8; ++e) afr[8 + e] = ap[kba0 + 8 + e];
#pragma unroll
            for (int in = 0; in < 2; ++in)
                acc[im][in] = __builtin_amdgcn_wmma_f32_16x16x32_bf16(
                    false, afr, false, bfr[in], (short)0, acc[im][in], false, false);
        }
        __syncthreads();
    }

    // ---- epilogue: bias + residual + relu; C/D: col=lane%16, rows r+(lane<16?0:8) ----
#pragma unroll
    for (int im = 0; im < 2; ++im)
#pragma unroll
        for (int in = 0; in < 2; ++in) {
            const int col = block_n + wn * 32 + in * 16 + (lane & 15);
            if (col >= N) continue;
            const int rbase = block_m + wm * 32 + im * 16 + ((lane < 16) ? 0 : 8);
            const float bv = bias ? bias[col] : 0.0f;
#pragma unroll
            for (int r = 0; r < 8; ++r) {
                const int row = rbase + r;
                if (row >= M) continue;
                float v = acc[im][in][r] + bv;
                if (residual) v += residual[(size_t)row * N + col];
                if (relu_flag) v = fmaxf(v, 0.0f);
                Cout[(size_t)row * N + col] = v;
            }
        }
}

// =====================================================================
// small helper kernels
// =====================================================================
__global__ void k_f2bf(const float* __restrict__ s, bf16_t* __restrict__ d, int n) {
    int i = blockIdx.x * blockDim.x + threadIdx.x;
    if (i < n) d[i] = (bf16_t)s[i];
}

// batched transpose-convert: mats of (rows x cols) f32 -> mats of (cols x rows) bf16
__global__ void k_f2bf_bt(const float* __restrict__ s, bf16_t* __restrict__ d,
                          int rows, int cols) {
    const size_t base = (size_t)blockIdx.y * rows * cols;
    int i = blockIdx.x * blockDim.x + threadIdx.x;
    if (i < rows * cols) {
        int r = i / cols, c = i % cols;
        d[base + (size_t)c * rows + r] = (bf16_t)s[base + i];
    }
}

__global__ void k_fill(float* __restrict__ p, float v, int n) {
    int i = blockIdx.x * blockDim.x + threadIdx.x;
    if (i < n) p[i] = v;
}

__global__ void k_sqrt_ip(float* __restrict__ p, int n) {
    int i = blockIdx.x * blockDim.x + threadIdx.x;
    if (i < n) p[i] = sqrtf(p[i]);
}

// column-degree accumulation: deg[b*C + dst] += w  (deg pre-filled with 1.0 = identity diag)
__global__ void k_edges(const int* __restrict__ hist, float* __restrict__ deg) {
    using namespace cfg;
    int i = blockIdx.x * blockDim.x + threadIdx.x;
    if (i >= EC) return;
    int t = i % (S - 1);
    int h = (i / (S - 1)) % HT;
    int b = i / ((S - 1) * HT);
    const int* row = hist + ((size_t)b * HT + h) * S;
    int s = row[t], d = row[t + 1];
    if (s != C && d != C && s != d) {
        float w = 1.0f + __expf(-0.01f * (float)(HT - 1 - h));
        atomicAdd(&deg[b * C + d], w);
    }
}

// Eg diag init: Eg[b,i,:] = d[b,i]^2 * E2[i,:]  (A' diagonal is identity)
__global__ void k_gcn_diag(const float* __restrict__ E2, const float* __restrict__ deg,
                           float* __restrict__ Eg) {
    using namespace cfg;
    size_t i = (size_t)blockIdx.x * blockDim.x + threadIdx.x;
    if (i >= (size_t)B * C * DIM) return;
    int dch = (int)(i % DIM);
    int c   = (int)((i / DIM) % C);
    int b   = (int)(i / ((size_t)DIM * C));
    float s = deg[b * C + c];
    Eg[i] = s * s * E2[(size_t)c * DIM + dch];
}

// sparse SpMM over the edge list: Eg[b,src,:] += d[src]*w*d[dst] * E2[dst,:]
__global__ __launch_bounds__(128)
void k_gcn_edges(const int* __restrict__ hist, const float* __restrict__ E2,
                 const float* __restrict__ deg, float* __restrict__ Eg) {
    using namespace cfg;
    int e   = blockIdx.x;       // 0..EC-1
    int dch = threadIdx.x;      // 0..DIM-1
    int t = e % (S - 1);
    int h = (e / (S - 1)) % HT;
    int b = e / ((S - 1) * HT);
    const int* row = hist + ((size_t)b * HT + h) * S;
    int s = row[t], d = row[t + 1];
    if (s == C || d == C || s == d) return;
    float w = 1.0f + __expf(-0.01f * (float)(HT - 1 - h));
    float coef = deg[b * C + s] * w * deg[b * C + d];
    atomicAdd(&Eg[((size_t)b * C + s) * DIM + dch], coef * E2[(size_t)d * DIM + dch]);
}

// pos[b,c,:] = concat(embedding[c,:], relu(Eg[b,c,:] + gcn_b))
__global__ void k_pos(const float* __restrict__ emb, const float* __restrict__ Eg,
                      const float* __restrict__ gcn_b, float* __restrict__ pos) {
    using namespace cfg;
    size_t i = (size_t)blockIdx.x * blockDim.x + threadIdx.x;
    if (i >= (size_t)B * C * D) return;
    int dch = (int)(i % D);
    int c   = (int)((i / D) % C);
    int b   = (int)(i / ((size_t)D * C));
    float v;
    if (dch < DIM) {
        v = emb[(size_t)c * DIM + dch];
    } else {
        int dd = dch - DIM;
        v = Eg[((size_t)b * C + c) * DIM + dd] + gcn_b[dd];
        v = fmaxf(v, 0.0f);
    }
    pos[i] = v;
}

// fused embedding + temporal encoding for all 136 sequences; also key-validity mask
__global__ void k_fuse(const int* __restrict__ hist, const int* __restrict__ cur,
                       const float* __restrict__ pos, const float* __restrict__ null_e,
                       float* __restrict__ X, unsigned char* __restrict__ keyvalid) {
    using namespace cfg;
    size_t i = (size_t)blockIdx.x * blockDim.x + threadIdx.x;
    if (i >= (size_t)ROWS * D) return;
    int dch = (int)(i % D);
    int s   = (int)((i / D) % S);
    int n   = (int)(i / ((size_t)D * S));
    int tok, b;
    if (n < B * HT) { b = n / HT; tok = hist[(size_t)n * S + s]; }
    else            { b = n - B * HT; tok = cur[(size_t)b * S + s]; }
    int pair = dch >> 1;
    float den = __powf(10000.0f, (2.0f * (float)pair) / (float)D);
    float arg = (float)s / den;
    float tpe = (dch & 1) ? __cosf(arg) : __sinf(arg);
    float v = (tok == C) ? (null_e[dch] + tpe)
                         : (pos[((size_t)b * C + tok) * D + dch] + tpe);
    X[i] = v;
    if (dch == 0) keyvalid[(size_t)n * S + s] = (tok != C) ? 1 : 0;
}

// row LayerNorm over D=256, one block per row
__global__ __launch_bounds__(256)
void k_layernorm(const float* __restrict__ in, float* __restrict__ out,
                 const float* __restrict__ gam, const float* __restrict__ bet) {
    using namespace cfg;
    __shared__ float red[256];
    int row = blockIdx.x, d = threadIdx.x;
    float v = in[(size_t)row * D + d];
    red[d] = v; __syncthreads();
    for (int off = 128; off > 0; off >>= 1) { if (d < off) red[d] += red[d + off]; __syncthreads(); }
    float mean = red[0] * (1.0f / D);
    __syncthreads();
    float cdev = v - mean;
    red[d] = cdev * cdev; __syncthreads();
    for (int off = 128; off > 0; off >>= 1) { if (d < off) red[d] += red[d + off]; __syncthreads(); }
    float var = red[0] * (1.0f / D);
    out[(size_t)row * D + d] = cdev * rsqrtf(var + 1e-5f) * gam[d] + bet[d];
}

// encoder self-attention: grid (NSEQ, NH), block 64 = 2 waves; key-mask from token validity
__global__ __launch_bounds__(64)
void k_attn_self(const float* __restrict__ Q, const float* __restrict__ K,
                 const float* __restrict__ V, const unsigned char* __restrict__ keyvalid,
                 float* __restrict__ O) {
    using namespace cfg;
    __shared__ float sK[S][HD + 1];
    __shared__ float sV[S][HD + 1];
    __shared__ float sP[S][S + 1];
    const int n = blockIdx.x, h = blockIdx.y, tq = threadIdx.x;
    const size_t rbase = (size_t)n * S;
    {
        const float* kp = K + (rbase + tq) * D + h * HD;
        const float* vp = V + (rbase + tq) * D + h * HD;
#pragma unroll 8
        for (int d = 0; d < HD; ++d) { sK[tq][d] = kp[d]; sV[tq][d] = vp[d]; }
    }
    __syncthreads();
    float q[HD];
    {
        const float* qp = Q + (rbase + tq) * D + h * HD;
#pragma unroll 8
        for (int d = 0; d < HD; ++d) q[d] = qp[d];
    }
    float mx = -1e30f;
    for (int j = 0; j < S; ++j) {
        float sc = 0.0f;
#pragma unroll 8
        for (int d = 0; d < HD; ++d) sc += q[d] * sK[j][d];
        sc *= 0.125f;                                  // 1/sqrt(64)
        if (!keyvalid[rbase + j]) sc = -1e30f;
        sP[tq][j] = sc;
        mx = fmaxf(mx, sc);
    }
    float sum = 0.0f;
    for (int j = 0; j < S; ++j) { float e = __expf(sP[tq][j] - mx); sP[tq][j] = e; sum += e; }
    float inv = 1.0f / sum;
    float* op = O + (rbase + tq) * D + h * HD;
    for (int d = 0; d < HD; ++d) {
        float o = 0.0f;
#pragma unroll 8
        for (int j = 0; j < S; ++j) o += sP[tq][j] * sV[j][d];
        op[d] = o * inv;
    }
}

// window cross-attention: q from mh (8192 rows), kv from mc (512 rows), |i-j|<=WIN mask
__global__ __launch_bounds__(64)
void k_attn_wca(const float* __restrict__ Q, const float* __restrict__ K,
                const float* __restrict__ V, float* __restrict__ O) {
    using namespace cfg;
    __shared__ float sK[S][HD + 1];
    __shared__ float sV[S][HD + 1];
    __shared__ float sP[S][S + 1];
    const int n = blockIdx.x, h = blockIdx.y, tq = threadIdx.x; // n in [0,128)
    const size_t qbase  = (size_t)n * S;
    const size_t kvbase = (size_t)(n / HT) * S;
    {
        const float* kp = K + (kvbase + tq) * D + h * HD;
        const float* vp = V + (kvbase + tq) * D + h * HD;
#pragma unroll 8
        for (int d = 0; d < HD; ++d) { sK[tq][d] = kp[d]; sV[tq][d] = vp[d]; }
    }
    __syncthreads();
    float q[HD];
    {
        const float* qp = Q + (qbase + tq) * D + h * HD;
#pragma unroll 8
        for (int d = 0; d < HD; ++d) q[d] = qp[d];
    }
    float mx = -1e30f;
    for (int j = 0; j < S; ++j) {
        float sc = 0.0f;
#pragma unroll 8
        for (int d = 0; d < HD; ++d) sc += q[d] * sK[j][d];
        sc *= 0.125f;
        int dd = tq - j; if (dd < 0) dd = -dd;
        if (dd > WIN) sc = -1e30f;
        sP[tq][j] = sc;
        mx = fmaxf(mx, sc);
    }
    float sum = 0.0f;
    for (int j = 0; j < S; ++j) { float e = __expf(sP[tq][j] - mx); sP[tq][j] = e; sum += e; }
    float inv = 1.0f / sum;
    float* op = O + (qbase + tq) * D + h * HD;
    for (int d = 0; d < HD; ++d) {
        float o = 0.0f;
#pragma unroll 8
        for (int j = 0; j < S; ++j) o += sP[tq][j] * sV[j][d];
        op[d] = o * inv;
    }
}

// trajectory aggregation: scores over 16 trajectories, softmax, weighted sum, +mc
// grid = B*S blocks, block = 256 (= D)
__global__ __launch_bounds__(256)
void k_agg(const float* __restrict__ wca, const float* __restrict__ mc, float* __restrict__ G) {
    using namespace cfg;
    __shared__ float sc[HT];
    __shared__ float red[256];
    const int bs = blockIdx.x;
    const int b = bs / S, s = bs % S;
    const int t = threadIdx.x;
    const int h = t >> 4, p = t & 15;
    const float* mrow = mc + ((size_t)b * S + s) * D;
    {
        const float* wrow = wca + (((size_t)(b * HT + h)) * S + s) * D;
        float part = 0.0f;
#pragma unroll
        for (int i = 0; i < 16; ++i) part += wrow[p * 16 + i] * mrow[p * 16 + i];
        red[t] = part;
    }
    __syncthreads();
    if (p < 8) red[t] += red[t + 8];  __syncthreads();
    if (p < 4) red[t] += red[t + 4];  __syncthreads();
    if (p < 2) red[t] += red[t + 2];  __syncthreads();
    if (p == 0) sc[h] = (red[t] + red[t + 1]) * 0.0625f;   // / sqrt(256)
    __syncthreads();
    if (t == 0) {
        float mx = -1e30f;
        for (int i = 0; i < HT; ++i) mx = fmaxf(mx, sc[i]);
        float sum = 0.0f;
        for (int i = 0; i < HT; ++i) { sc[i] = __expf(sc[i] - mx); sum += sc[i]; }
        float inv = 1.0f / sum;
        for (int i = 0; i < HT; ++i) sc[i] *= inv;
    }
    __syncthreads();
    float a = 0.0f;
#pragma unroll
    for (int h2 = 0; h2 < HT; ++h2)
        a += sc[h2] * wca[(((size_t)(b * HT + h2)) * S + s) * D + t];
    G[((size_t)b * S + s) * D + t] = a + mrow[t];
}

// =====================================================================
// host side
// =====================================================================
static inline void gemm(hipStream_t st, const bf16_t* A, const bf16_t* Bt, float* Cm,
                        const float* bias, const float* resid,
                        int M, int N, int K, int relu) {
    dim3 g((N + GEMM_BN - 1) / GEMM_BN, (M + GEMM_BM - 1) / GEMM_BM);
    k_gemm_bf16<<<g, dim3(128), 0, st>>>(A, Bt, Cm, bias, resid, M, N, K, relu);
}

extern "C" void kernel_launch(void* const* d_in, const int* in_sizes, int n_in,
                              void* d_out, int out_size, void* d_ws, size_t ws_size,
                              hipStream_t stream) {
    (void)in_sizes; (void)n_in; (void)out_size; (void)ws_size;
    using namespace cfg;

    const int*   history = (const int*)d_in[0];
    const int*   cur     = (const int*)d_in[1];
    const float* emb     = (const float*)d_in[2];
    const float* null_e  = (const float*)d_in[3];
    const float* gcn_w   = (const float*)d_in[4];
    const float* gcn_b   = (const float*)d_in[5];
    const float* attn_w  = (const float*)d_in[6];
    const float* attn_b  = (const float*)d_in[7];
    const float* ffn_w1  = (const float*)d_in[8];
    const float* ffn_b1  = (const float*)d_in[9];
    const float* ffn_w2  = (const float*)d_in[10];
    const float* ffn_b2  = (const float*)d_in[11];
    const float* ln_s    = (const float*)d_in[12];
    const float* ln_b    = (const float*)d_in[13];
    const float* wca_w   = (const float*)d_in[14];
    const float* wca_b   = (const float*)d_in[15];
    const float* ial_w   = (const float*)d_in[16];
    const float* ial_b   = (const float*)d_in[17];
    float* out = (float*)d_out;

    // ---------- workspace carve (f32 arena, then bf16 arena, then bytes) ----------
    const size_t nDeg = (size_t)B * C;
    const size_t nE2  = (size_t)C * DIM;
    const size_t nEg  = (size_t)B * C * DIM;
    const size_t nPos = (size_t)B * C * D;
    const size_t nX   = (size_t)ROWS * D;
    const size_t nH   = (size_t)ROWS * DFF;
    const size_t nU   = (3 * nX > nH) ? 3 * nX : nH;  // Q/K/V union with H
    const size_t nG   = (size_t)B * S * D;
    const size_t nF   = (size_t)B * S * DIM;

    float* fp = (float*)d_ws;
    size_t o = 0;
    float* deg = fp + o; o += nDeg;
    float* E2  = fp + o; o += nE2;
    float* Eg  = fp + o; o += nEg;
    float* pos = fp + o; o += nPos;
    float* X   = fp + o; o += nX;
    float* T   = fp + o; o += nX;     // pre-LN sums / wca result
    float* U   = fp + o; o += nU;     // Q,K,V  (and H reuses U)
    float* O   = fp + o; o += nX;
    float* G   = fp + o; o += nG;
    float* Fm  = fp + o; o += nF;
    float* Qf = U, *Kf = U + nX, *Vf = U + 2 * nX, *Hf = U;

    bf16_t* bp = (bf16_t*)(fp + o);
    size_t ob = 0;
    bf16_t* wAttnB = bp + ob; ob += (size_t)L * 4 * D * D;   // transposed per matrix
    bf16_t* wF1B   = bp + ob; ob += (size_t)L * D * DFF;     // (DFF x D) per layer
    bf16_t* wF2B   = bp + ob; ob += (size_t)L * DFF * D;     // (D x DFF) per layer
    bf16_t* wWcaB  = bp + ob; ob += (size_t)4 * D * D;       // transposed per matrix
    bf16_t* wIalB  = bp + ob; ob += (size_t)D * DIM;         // (DIM x D)
    bf16_t* wGcnB  = bp + ob; ob += (size_t)DIM * DIM;       // (DIM x DIM) transposed
    bf16_t* embB   = bp + ob; ob += (size_t)C * DIM;         // (C x DIM): A of E2, Bt of logits
    bf16_t* XB     = bp + ob; ob += nX;
    bf16_t* OHB    = bp + ob; ob += nU;   // bf16 of O (attn) or H (ffn)
    bf16_t* GB     = bp + ob; ob += nG;
    bf16_t* FB     = bp + ob; ob += nF;
    unsigned char* keyvalid = (unsigned char*)(bp + ob);

    auto conv = [&](const float* s, bf16_t* d, size_t n) {
        int nn = (int)n;
        k_f2bf<<<(nn + 255) / 256, 256, 0, stream>>>(s, d, nn);
    };
    auto convT = [&](const float* s, bf16_t* d, int mats, int rows, int cols) {
        dim3 g((rows * cols + 255) / 256, mats);
        k_f2bf_bt<<<g, 256, 0, stream>>>(s, d, rows, cols);
    };

    // ---------- weight conversions (pre-transposed to N x K) ----------
    convT(attn_w, wAttnB, L * 4, D, D);
    convT(ffn_w1, wF1B,   L,     D, DFF);
    convT(ffn_w2, wF2B,   L,     DFF, D);
    convT(wca_w,  wWcaB,  4,     D, D);
    convT(ial_w,  wIalB,  1,     D, DIM);
    convT(gcn_w,  wGcnB,  1,     DIM, DIM);
    conv(emb, embB, (size_t)C * DIM);

    // ---------- graph -> normalized sparse GCN ----------
    k_fill<<<((int)nDeg + 255) / 256, 256, 0, stream>>>(deg, 1.0f, (int)nDeg);
    k_edges<<<(EC + 255) / 256, 256, 0, stream>>>(history, deg);
    k_sqrt_ip<<<((int)nDeg + 255) / 256, 256, 0, stream>>>(deg, (int)nDeg);
    // E2 = embedding @ gcn_w   (A@E@W == A@(E@W))
    gemm(stream, embB, wGcnB, E2, nullptr, nullptr, C, DIM, DIM, 0);
    k_gcn_diag<<<(int)((nEg + 255) / 256), 256, 0, stream>>>(E2, deg, Eg);
    k_gcn_edges<<<EC, DIM, 0, stream>>>(history, E2, deg, Eg);
    k_pos<<<(int)((nPos + 255) / 256), 256, 0, stream>>>(emb, Eg, gcn_b, pos);

    // ---------- fuse (history seqs + current seqs into one batch) ----------
    k_fuse<<<(int)((nX + 255) / 256), 256, 0, stream>>>(history, cur, pos, null_e, X, keyvalid);

    // ---------- shared encoder over 136 sequences ----------
    for (int l = 0; l < L; ++l) {
        conv(X, XB, nX);
        const bf16_t* wl = wAttnB + (size_t)l * 4 * D * D;
        const float*  bl = attn_b + (size_t)l * 4 * D;
        gemm(stream, XB, wl + 0 * D * D, Qf, bl + 0 * D, nullptr, ROWS, D, D, 0);
        gemm(stream, XB, wl + 1 * D * D, Kf, bl + 1 * D, nullptr, ROWS, D, D, 0);
        gemm(stream, XB, wl + 2 * D * D, Vf, bl + 2 * D, nullptr, ROWS, D, D, 0);
        k_attn_self<<<dim3(NSEQ, NH), 64, 0, stream>>>(Qf, Kf, Vf, keyvalid, O);
        conv(O, OHB, nX);
        gemm(stream, OHB, wl + 3 * D * D, T, bl + 3 * D, /*residual=*/X, ROWS, D, D, 0);
        k_layernorm<<<ROWS, 256, 0, stream>>>(T, X, ln_s + (size_t)(l * 2 + 0) * D,
                                                    ln_b + (size_t)(l * 2 + 0) * D);
        conv(X, XB, nX);
        gemm(stream, XB, wF1B + (size_t)l * D * DFF, Hf, ffn_b1 + (size_t)l * DFF,
             nullptr, ROWS, DFF, D, 1);
        conv(Hf, OHB, nH);
        gemm(stream, OHB, wF2B + (size_t)l * DFF * D, T, ffn_b2 + (size_t)l * D,
             /*residual=*/X, ROWS, D, DFF, 0);
        k_layernorm<<<ROWS, 256, 0, stream>>>(T, X, ln_s + (size_t)(l * 2 + 1) * D,
                                                    ln_b + (size_t)(l * 2 + 1) * D);
    }

    // ---------- window cross-attention (q = mh, kv = mc replicated per trajectory) ----------
    const int MH = B * HT * S;  // 8192
    const int MC = B * S;       // 512
    conv(X, XB, nX);
    const float*  mc  = X  + (size_t)(B * HT) * S * D;
    const bf16_t* mhB = XB;
    const bf16_t* mcB = XB + (size_t)(B * HT) * S * D;
    gemm(stream, mhB, wWcaB + 0 * D * D, Qf, wca_b + 0 * D, nullptr, MH, D, D, 0);
    gemm(stream, mcB, wWcaB + 1 * D * D, Kf, wca_b + 1 * D, nullptr, MC, D, D, 0);
    gemm(stream, mcB, wWcaB + 2 * D * D, Vf, wca_b + 2 * D, nullptr, MC, D, D, 0);
    k_attn_wca<<<dim3(B * HT, NH), 64, 0, stream>>>(Qf, Kf, Vf, O);
    conv(O, OHB, (size_t)MH * D);
    gemm(stream, OHB, wWcaB + 3 * D * D, T, wca_b + 3 * D, /*residual(mh)=*/X, MH, D, D, 0);
    // T = wca, shape (B,HT,S,D)

    // ---------- trajectory aggregation + IAL + output projection ----------
    k_agg<<<B * S, 256, 0, stream>>>(T, mc, G);
    conv(G, GB, nG);
    gemm(stream, GB, wIalB, Fm, ial_b, nullptr, B * S, DIM, D, 1);   // relu
    conv(Fm, FB, nF);
    gemm(stream, FB, embB, out, nullptr, nullptr, B * S, C, DIM, 0);
}
